// SNNPhaseEncoder_32547262169272
// MI455X (gfx1250) — matrix operand order repaired and checked by reading, not compiled, not run on hardware
//
#include <hip/hip_runtime.h>

#define BATCH 32
#define TIME  512
#define INF   512
#define HIDN  1024
#define BT    (BATCH * TIME)   // GEMM M = 16384

typedef __attribute__((ext_vector_type(16))) _Float16 v16h;
typedef __attribute__((ext_vector_type(8)))  _Float16 v8h;
typedef __attribute__((ext_vector_type(4)))  _Float16 v4h;
typedef __attribute__((ext_vector_type(8)))  float    v8f;
typedef __attribute__((ext_vector_type(4)))  float    v4f;

// ---------------------------------------------------------------------------
// Pass 0: split f32 -> (hi, lo) f16 arrays, one-time.
// ---------------------------------------------------------------------------
__global__ __launch_bounds__(256) void split_f32_to_f16(const float* __restrict__ src,
                                                        _Float16* __restrict__ hi,
                                                        _Float16* __restrict__ lo,
                                                        int n4) {
    int i = blockIdx.x * blockDim.x + threadIdx.x;
    if (i >= n4) return;
    v4f v = ((const v4f*)src)[i];
    v4h h, l;
#pragma unroll
    for (int j = 0; j < 4; ++j) {
        _Float16 hj = (_Float16)v[j];
        h[j] = hj;
        l[j] = (_Float16)(v[j] - (float)hj);
    }
    ((v4h*)hi)[i] = h;
    ((v4h*)lo)[i] = l;
}

__device__ __forceinline__ v16h join8(v8h a, v8h b) {
    return __builtin_shufflevector(a, b, 0, 1, 2, 3, 4, 5, 6, 7,
                                         8, 9, 10, 11, 12, 13, 14, 15);
}

// ---------------------------------------------------------------------------
// Main GEMM (pre-split f16 operands): hidden[m,n] = sum_k x[m,k]*W[n,k] + b[n]
// One wave owns a 64x32 tile = 4x2 WMMA tiles; split-f16 hi*hi + hi*lo + lo*hi.
// Addressing: uniform 64-bit base (SADDR form) + loop-invariant 32-bit lane
// offsets; k0 folds into the load immediate via unrolling -> no VALU address
// math in the K loop and fewer WMMA->VALU hazard NOPs.
// ---------------------------------------------------------------------------
__global__ __launch_bounds__(256) void snn_gemm_wmma_f16(const _Float16* __restrict__ Xhi,
                                                         const _Float16* __restrict__ Xlo,
                                                         const _Float16* __restrict__ Whi,
                                                         const _Float16* __restrict__ Wlo,
                                                         const float* __restrict__ Bv,
                                                         float* __restrict__ H) {
    const int lane = threadIdx.x & 31;
    const int wavesPerBlk = blockDim.x >> 5;
    const int wid = blockIdx.x * wavesPerBlk + (threadIdx.x >> 5);

    const int N_TILES = HIDN / 32;          // 32 wave tiles along N
    const int tm = wid / N_TILES;
    const int tn = wid - tm * N_TILES;
    const int m0 = tm * 64;
    const int n0 = tn * 32;
    const int r  = lane & 15;
    const int g  = lane >> 4;

    // Loop-invariant 32-bit per-lane offsets (elements).
    // A fragment: lane holds row m; elems 0-7 -> K = 8g+0..7, elems 8-15 -> K = 16+8g+0..7
    uint32_t offA[4];
#pragma unroll
    for (int ms = 0; ms < 4; ++ms)
        offA[ms] = (uint32_t)(m0 + ms * 16 + r) * INF + 8u * (uint32_t)g;
    // B fragment: lane holds col n; elems 0-7 -> K = 8g+0..7, elems 8-15 -> K = 16+8g+0..7
    uint32_t offB[2];
#pragma unroll
    for (int ns = 0; ns < 2; ++ns)
        offB[ns] = (uint32_t)(n0 + ns * 16 + r) * INF + 8u * (uint32_t)g;

    v8f acc[4][2] = {};

#pragma unroll 2
    for (uint32_t k0 = 0; k0 < INF; k0 += 32) {
        v16h ah[4], al[4], bh[2], bl[2];
#pragma unroll
        for (int ms = 0; ms < 4; ++ms) {
            v8h h0 = *(const v8h*)(Xhi + (offA[ms] + k0));
            v8h h1 = *(const v8h*)(Xhi + (offA[ms] + k0 + 16));
            ah[ms] = join8(h0, h1);
            v8h l0 = *(const v8h*)(Xlo + (offA[ms] + k0));
            v8h l1 = *(const v8h*)(Xlo + (offA[ms] + k0 + 16));
            al[ms] = join8(l0, l1);
        }
#pragma unroll
        for (int ns = 0; ns < 2; ++ns) {
            v8h h0 = *(const v8h*)(Whi + (offB[ns] + k0));
            v8h h1 = *(const v8h*)(Whi + (offB[ns] + k0 + 16));
            bh[ns] = join8(h0, h1);
            v8h l0 = *(const v8h*)(Wlo + (offB[ns] + k0));
            v8h l1 = *(const v8h*)(Wlo + (offB[ns] + k0 + 16));
            bl[ns] = join8(l0, l1);
        }
#pragma unroll
        for (int ms = 0; ms < 4; ++ms) {
#pragma unroll
            for (int ns = 0; ns < 2; ++ns) {
                acc[ms][ns] = __builtin_amdgcn_wmma_f32_16x16x32_f16(
                    false, ah[ms], false, bh[ns], (short)0, acc[ms][ns], false, false);
                acc[ms][ns] = __builtin_amdgcn_wmma_f32_16x16x32_f16(
                    false, ah[ms], false, bl[ns], (short)0, acc[ms][ns], false, false);
                acc[ms][ns] = __builtin_amdgcn_wmma_f32_16x16x32_f16(
                    false, al[ms], false, bh[ns], (short)0, acc[ms][ns], false, false);
            }
        }
    }

    // Epilogue: C/D layout -> element e in lane l is (M = e + 8*(l/16), N = l%16)
#pragma unroll
    for (int ns = 0; ns < 2; ++ns) {
        float bv = Bv[n0 + ns * 16 + r];
#pragma unroll
        for (int ms = 0; ms < 4; ++ms) {
            float* outp = H + (size_t)(m0 + ms * 16 + 8 * g) * HIDN + (n0 + ns * 16 + r);
#pragma unroll
            for (int e = 0; e < 8; ++e)
                outp[(size_t)e * HIDN] = acc[ms][ns][e] + bv;
        }
    }
}

// ---------------------------------------------------------------------------
// Fallback GEMM (conversion fused in-loop) used only if d_ws is too small.
// ---------------------------------------------------------------------------
__device__ __forceinline__ void split8(const float* __restrict__ p, v16h& hi, v16h& lo, int e0) {
    v4f f0 = *(const v4f*)(p);
    v4f f1 = *(const v4f*)(p + 4);
#pragma unroll
    for (int j = 0; j < 4; ++j) {
        float a  = f0[j];
        _Float16 h = (_Float16)a;
        hi[e0 + j]     = h;
        lo[e0 + j]     = (_Float16)(a - (float)h);
        float c  = f1[j];
        _Float16 h2 = (_Float16)c;
        hi[e0 + 4 + j] = h2;
        lo[e0 + 4 + j] = (_Float16)(c - (float)h2);
    }
}

__global__ __launch_bounds__(256) void snn_gemm_wmma_fused(const float* __restrict__ X,
                                                           const float* __restrict__ W,
                                                           const float* __restrict__ Bv,
                                                           float* __restrict__ H) {
    const int lane = threadIdx.x & 31;
    const int wavesPerBlk = blockDim.x >> 5;
    const int wid = blockIdx.x * wavesPerBlk + (threadIdx.x >> 5);

    const int N_TILES = HIDN / 32;
    const int tm = wid / N_TILES;
    const int tn = wid - tm * N_TILES;
    const int m0 = tm * 64;
    const int n0 = tn * 32;
    const int r  = lane & 15;
    const int g  = lane >> 4;

    const float* xrow[4];
#pragma unroll
    for (int ms = 0; ms < 4; ++ms)
        xrow[ms] = X + (size_t)(m0 + ms * 16 + r) * INF;
    const float* wrow[2];
#pragma unroll
    for (int ns = 0; ns < 2; ++ns)
        wrow[ns] = W + (size_t)(n0 + ns * 16 + r) * INF;

    v8f acc[4][2] = {};

    for (int k0 = 0; k0 < INF; k0 += 32) {
        v16h ah[4], al[4], bh[2], bl[2];
#pragma unroll
        for (int ms = 0; ms < 4; ++ms) {
            split8(xrow[ms] + k0 + 8 * g,      ah[ms], al[ms], 0);
            split8(xrow[ms] + k0 + 16 + 8 * g, ah[ms], al[ms], 8);
        }
#pragma unroll
        for (int ns = 0; ns < 2; ++ns) {
            split8(wrow[ns] + k0 + 8 * g,      bh[ns], bl[ns], 0);
            split8(wrow[ns] + k0 + 16 + 8 * g, bh[ns], bl[ns], 8);
        }
#pragma unroll
        for (int ms = 0; ms < 4; ++ms) {
#pragma unroll
            for (int ns = 0; ns < 2; ++ns) {
                acc[ms][ns] = __builtin_amdgcn_wmma_f32_16x16x32_f16(
                    false, ah[ms], false, bh[ns], (short)0, acc[ms][ns], false, false);
                acc[ms][ns] = __builtin_amdgcn_wmma_f32_16x16x32_f16(
                    false, ah[ms], false, bl[ns], (short)0, acc[ms][ns], false, false);
                acc[ms][ns] = __builtin_amdgcn_wmma_f32_16x16x32_f16(
                    false, al[ms], false, bh[ns], (short)0, acc[ms][ns], false, false);
            }
        }
    }

#pragma unroll
    for (int ns = 0; ns < 2; ++ns) {
        float bv = Bv[n0 + ns * 16 + r];
#pragma unroll
        for (int ms = 0; ms < 4; ++ms) {
            float* outp = H + (size_t)(m0 + ms * 16 + 8 * g) * HIDN + (n0 + ns * 16 + r);
#pragma unroll
            for (int e = 0; e < 8; ++e)
                outp[(size_t)e * HIDN] = acc[ms][ns][e] + bv;
        }
    }
}

// ---------------------------------------------------------------------------
// In-place LIF scan over time: mem = 0.5*mem + h_t; spike if mem > 1; hard reset.
// ---------------------------------------------------------------------------
__global__ __launch_bounds__(256) void snn_scan(float* __restrict__ H) {
    int idx = blockIdx.x * blockDim.x + threadIdx.x;
    if (idx >= BATCH * HIDN) return;
    int b = idx / HIDN;
    int h = idx - b * HIDN;
    float* p = H + (size_t)b * TIME * HIDN + h;
    float mem = 0.0f;
#pragma unroll 8
    for (int t = 0; t < TIME; ++t) {
        float v = p[(size_t)t * HIDN];
        mem = 0.5f * mem + v;
        bool s = mem > 1.0f;
        p[(size_t)t * HIDN] = s ? 1.0f : 0.0f;
        mem = s ? 0.0f : mem;
    }
}

extern "C" void kernel_launch(void* const* d_in, const int* in_sizes, int n_in,
                              void* d_out, int out_size, void* d_ws, size_t ws_size,
                              hipStream_t stream) {
    (void)in_sizes; (void)n_in; (void)out_size;
    const float* x  = (const float*)d_in[0];   // [32, 512, 512]
    const float* W  = (const float*)d_in[1];   // [1024, 512]
    const float* bv = (const float*)d_in[2];   // [1024]
    float* out = (float*)d_out;                // [32, 512, 1024]

    const int waveTiles = (BT / 64) * (HIDN / 32);   // 8192 wave tiles
    const int gemmBlocks = waveTiles / 8;            // 1024 blocks of 256 thr

    const size_t xcnt = (size_t)BT * INF;            // 8,388,608
    const size_t wcnt = (size_t)HIDN * INF;          //   524,288
    const size_t need = (2 * xcnt + 2 * wcnt) * sizeof(_Float16);  // ~34 MB

    if (ws_size >= need) {
        _Float16* Xhi = (_Float16*)d_ws;
        _Float16* Xlo = Xhi + xcnt;
        _Float16* Whi = Xlo + xcnt;
        _Float16* Wlo = Whi + wcnt;

        const int xn4 = (int)(xcnt / 4);
        const int wn4 = (int)(wcnt / 4);
        split_f32_to_f16<<<dim3((xn4 + 255) / 256), dim3(256), 0, stream>>>(x, Xhi, Xlo, xn4);
        split_f32_to_f16<<<dim3((wn4 + 255) / 256), dim3(256), 0, stream>>>(W, Whi, Wlo, wn4);

        snn_gemm_wmma_f16<<<dim3(gemmBlocks), dim3(256), 0, stream>>>(Xhi, Xlo, Whi, Wlo, bv, out);
    } else {
        snn_gemm_wmma_fused<<<dim3(gemmBlocks), dim3(256), 0, stream>>>(x, W, bv, out);
    }

    const int chains = BATCH * HIDN;                 // 32768
    snn_scan<<<dim3((chains + 255) / 256), dim3(256), 0, stream>>>(out);
}
